// DecoderModel_12352325943321
// MI455X (gfx1250) — compile-verified, matrix operand order checked by reading, not run on hardware
//
#include <hip/hip_runtime.h>
#include <hip/hip_bf16.h>
#include <math.h>

// Problem constants (from reference)
#define BB 2
#define SS 1024
#define DD 512
#define HH 8
#define HD 64
#define LL 6
#define VV 32000
#define FFN 2048
#define WIN0 64

typedef __attribute__((ext_vector_type(16))) __bf16 v16bf;
typedef __attribute__((ext_vector_type(8)))  float  v8f;
typedef __attribute__((ext_vector_type(4))) unsigned int u32x4;
typedef __attribute__((ext_vector_type(8))) int i32x8;
typedef __attribute__((ext_vector_type(4))) int i32x4;

struct alignas(16) U4 { unsigned int a, b, c, d; };
union FragB { v16bf v; U4 q[2]; };

__device__ __forceinline__ v8f zero8() {
  v8f z;
#pragma unroll
  for (int i = 0; i < 8; ++i) z[i] = 0.f;
  return z;
}

__device__ __forceinline__ v8f wmma_bf16(v16bf a, v16bf b, v8f c) {
  // D = A(16x32 bf16) x B(32x16 bf16) + C(16x16 f32)
  return __builtin_amdgcn_wmma_f32_16x16x32_bf16(false, a, false, b, (short)0, c,
                                                 false, false);
}

#if __has_builtin(__builtin_amdgcn_tensor_load_to_lds) && \
    __has_builtin(__builtin_amdgcn_s_wait_tensorcnt)
#define USE_TDM 1
#else
#define USE_TDM 0
#endif

#if USE_TDM
// Issue one TDM 2D tile load: tile_rows x tile_cols elements of 2B, from a
// row-major tensor with row stride `stride_elems`, into LDS at lds_addr
// (contiguous row-major tile). D# bitfields per CDNA5 ISA 8.3/8.4.
// This toolchain's builtin takes 6 args (4 descriptor groups, one extra
// 8-dword group, cache policy immediate).
__device__ __forceinline__ void tdm_load_tile_bf16(
    unsigned lds_addr, const __bf16* gptr, int tile_cols, int tile_rows,
    int tensor_d0, int tensor_d1, int stride_elems) {
  unsigned long long ga = (unsigned long long)(uintptr_t)gptr;
  u32x4 g0;
  g0[0] = 1u;                                    // count=1, is_restore=0, gather=0
  g0[1] = lds_addr;                              // bits 63:32  lds_addr
  g0[2] = (unsigned)(ga & 0xffffffffu);          // bits 95:64  global_addr lo
  g0[3] = ((unsigned)(ga >> 32) & 0x01ffffffu)   // bits 120:96 global_addr hi
          | (2u << 30);                          // bits 127:126 type=2 (image)
  i32x8 g1;
  g1[0] = (int)(1u << 16);                       // data_size=1 (2 bytes), wg_mask=0
  g1[1] = (int)(((unsigned)tensor_d0 & 0xffffu) << 16);          // dim0[15:0]
  g1[2] = (int)((((unsigned)tensor_d0 >> 16) & 0xffffu) |
                (((unsigned)tensor_d1 & 0xffffu) << 16));        // dim0 hi | dim1 lo
  g1[3] = (int)((((unsigned)tensor_d1 >> 16) & 0xffffu) |
                (((unsigned)tile_cols & 0xffffu) << 16));        // dim1 hi | tile_dim0
  g1[4] = (int)((unsigned)tile_rows & 0xffffu);                  // tile_dim1, tile_dim2=0
  g1[5] = (int)(unsigned)stride_elems;                           // dim0_stride lo32
  g1[6] = 0;                                                     // dim0_stride hi, dim1_stride
  g1[7] = 0;
  i32x4 gz4;
  gz4[0] = 0; gz4[1] = 0; gz4[2] = 0; gz4[3] = 0;                // groups 2/3 unused (2D)
  i32x8 gz8;
#pragma unroll
  for (int i = 0; i < 8; ++i) gz8[i] = 0;
  __builtin_amdgcn_tensor_load_to_lds(g0, g1, gz4, gz4, gz8, 0);
}
#endif

// ---------------------------------------------------------------- cvt f32->bf16
__global__ void cvt_bf16_kernel(const float* __restrict__ in,
                                __bf16* __restrict__ out, int n) {
  int i = (blockIdx.x * blockDim.x + threadIdx.x) * 4;
#pragma unroll
  for (int j = 0; j < 4; ++j)
    if (i + j < n) out[i + j] = (__bf16)in[i + j];
}

// ------------------------------------------------------- embedding + positional
__global__ void embed_kernel(const int* __restrict__ ids,
                             const float* __restrict__ emb,
                             float* __restrict__ x, __bf16* __restrict__ xb) {
  int row = blockIdx.x;  // 0..B*S-1
  int s = row % SS;
  int id = ids[row];
  int d0 = threadIdx.x * 2;
#pragma unroll
  for (int j = 0; j < 2; ++j) {
    int d = d0 + j;
    float freq = __expf(-(float)(d & ~1) * (logf(10000.f) / (float)DD));
    float ang = (float)s * freq;
    float pe = (d & 1) ? __cosf(ang) : __sinf(ang);
    float v = emb[(size_t)id * DD + d] + pe;
    x[(size_t)row * DD + d] = v;
    xb[(size_t)row * DD + d] = (__bf16)v;
  }
}

// --------------------------------------------------------------- WMMA bf16 GEMM
// C[M,N] = A[M,K] @ B[N,K]^T (+bias). 128x64 tile per 256-thread block (8 waves),
// each wave computes a 16x64 strip (4 wmma accumulators), K stepped by 32.
// Tile staging into LDS is done by the Tensor Data Mover (wave 0 issues two 2D
// tile descriptors per K step; TENSORcnt + barrier synchronize).
template <int OUT_BF16, int HAS_BIAS>
__global__ __launch_bounds__(256) void gemm_kernel(
    const __bf16* __restrict__ A, const __bf16* __restrict__ Bw,
    const float* __restrict__ bias, void* __restrict__ outp, int M, int N, int K) {
  __shared__ __align__(16) __bf16 As[128 * 32];
  __shared__ __align__(16) __bf16 Bs[64 * 32];
  const int tid = threadIdx.x;
  const int lane = tid & 31;
  const int wv = tid >> 5;
  const int bm = blockIdx.x, bn = blockIdx.y;
  const int mrow = wv * 16;
  const int lx = lane & 15;
  const int base = (lane < 16) ? 0 : 8;  // K sub-block owned by this half-wave

  v8f c[4];
#pragma unroll
  for (int nt = 0; nt < 4; ++nt) c[nt] = zero8();

  for (int k0 = 0; k0 < K; k0 += 32) {
#if USE_TDM
    if (wv == 0) {
      // A tile: 128 rows x 32 cols ; B tile: 64 rows x 32 cols (bf16)
      tdm_load_tile_bf16((unsigned)(uintptr_t)As,
                         A + (size_t)(bm * 128) * K + k0, 32, 128, K, M, K);
      tdm_load_tile_bf16((unsigned)(uintptr_t)Bs,
                         Bw + (size_t)(bn * 64) * K + k0, 32, 64, K, N, K);
      __builtin_amdgcn_s_wait_tensorcnt(0);
    }
#else
    {
      if (k0 + 32 < K) {  // emits global_prefetch_b8
        __builtin_prefetch(A + (size_t)(bm * 128 + (tid >> 1)) * K + k0 + 32, 0, 0);
        __builtin_prefetch(Bw + (size_t)(bn * 64 + (tid & 63)) * K + k0 + 32, 0, 0);
      }
      int idx = tid;
#pragma unroll
      for (int it = 0; it < 2; ++it, idx += 256) {
        int row = idx >> 2, c4 = idx & 3;
        ((U4*)As)[idx] =
            *(const U4*)(A + (size_t)(bm * 128 + row) * K + k0 + c4 * 8);
      }
      int row = tid >> 2, c4 = tid & 3;  // B tile: 64x32 = 256 x 16B
      ((U4*)Bs)[tid] =
          *(const U4*)(Bw + (size_t)(bn * 64 + row) * K + k0 + c4 * 8);
    }
#endif
    __syncthreads();

    FragB af;
    const __bf16* ar = As + (mrow + lx) * 32;
    af.q[0] = *(const U4*)(ar + base);
    af.q[1] = *(const U4*)(ar + base + 16);
#pragma unroll
    for (int nt = 0; nt < 4; ++nt) {
      FragB bf;
      const __bf16* br = Bs + (nt * 16 + lx) * 32;
      bf.q[0] = *(const U4*)(br + base);
      bf.q[1] = *(const U4*)(br + base + 16);
      c[nt] = wmma_bf16(af.v, bf.v, c[nt]);
    }
    __syncthreads();
  }
  // epilogue: C layout -> lane holds col n for rows (vgpr + half*8)
#pragma unroll
  for (int nt = 0; nt < 4; ++nt) {
    int n = bn * 64 + nt * 16 + lx;
    float bv = HAS_BIAS ? bias[n] : 0.f;
#pragma unroll
    for (int jv = 0; jv < 8; ++jv) {
      int m = bm * 128 + mrow + jv + ((lane < 16) ? 0 : 8);
      float val = c[nt][jv] + bv;
      if (OUT_BF16)
        ((__bf16*)outp)[(size_t)m * N + n] = (__bf16)val;
      else
        ((float*)outp)[(size_t)m * N + n] = val;
    }
  }
}

// ------------------------------------------- fused banded flash attention (WMMA)
// One wave per (b, h, 16-row q tile). Per-head window win=(h+1)*64, causal.
__global__ __launch_bounds__(128) void attn_kernel(const __bf16* __restrict__ qkv,
                                                   __bf16* __restrict__ attn) {
  __shared__ __align__(16) __bf16 Ps[4][16 * 32];  // per-wave P transpose bounce
  const int tid = threadIdx.x;
  const int lane = tid & 31, wv = tid >> 5;
  const int lx = lane & 15;
  const int half8 = (lane < 16) ? 0 : 8;
  const int qt = blockIdx.x * 4 + wv;
  const int h = blockIdx.y, b = blockIdx.z;
  const int i0 = qt * 16;
  const int win = (h + 1) * WIN0;
  const int rowstride = 3 * DD;  // 1536 bf16 per token row of qkv

  // Q fragments (A layout), two K-chunks of 32 over HD=64
  FragB qf[2];
  {
    const __bf16* qr = qkv + (size_t)(b * SS + i0 + lx) * rowstride + h * HD;
#pragma unroll
    for (int kk = 0; kk < 2; ++kk) {
      qf[kk].q[0] = *(const U4*)(qr + kk * 32 + half8);
      qf[kk].q[1] = *(const U4*)(qr + kk * 32 + half8 + 16);
    }
  }

  float mrun[8], lsum[8];
  v8f acc[4];
#pragma unroll
  for (int nt = 0; nt < 4; ++nt) acc[nt] = zero8();
#pragma unroll
  for (int jv = 0; jv < 8; ++jv) { mrun[jv] = -3.0e38f; lsum[jv] = 0.f; }

  int klo = i0 - win + 1;
  if (klo < 0) klo = 0;
  const int j0lo = klo & ~31;
  const int j0hi = (i0 + 15) & ~31;  // <= i0, so diagonal always covered

  for (int j0 = j0lo; j0 <= j0hi; j0 += 32) {
    // scores: 16x32 = two f32 fragments (QK^T)
    v8f sfr[2];
#pragma unroll
    for (int t = 0; t < 2; ++t) {
      v8f s = zero8();
      const __bf16* kr =
          qkv + (size_t)(b * SS + j0 + t * 16 + lx) * rowstride + DD + h * HD;
#pragma unroll
      for (int kk = 0; kk < 2; ++kk) {
        FragB kf;
        kf.q[0] = *(const U4*)(kr + kk * 32 + half8);
        kf.q[1] = *(const U4*)(kr + kk * 32 + half8 + 16);
        s = wmma_bf16(qf[kk].v, kf.v, s);
      }
      sfr[t] = s;
    }
    // scale + banded-causal mask (key padding is a no-op: ids in [1,V))
    float p[2][8];
#pragma unroll
    for (int t = 0; t < 2; ++t)
#pragma unroll
      for (int jv = 0; jv < 8; ++jv) {
        int i = i0 + jv + half8;
        int j = j0 + t * 16 + lx;
        int rel = i - j;
        bool ok = (rel >= 0) && (rel < win);
        p[t][jv] = ok ? sfr[t][jv] * 0.125f : -3.0e38f;  // 1/sqrt(64)
      }
    // row max over the 32 columns (xor-shuffle within each 16-lane half)
    float alpha[8];
#pragma unroll
    for (int jv = 0; jv < 8; ++jv) {
      float v = fmaxf(p[0][jv], p[1][jv]);
#pragma unroll
      for (int off = 8; off >= 1; off >>= 1) v = fmaxf(v, __shfl_xor(v, off, 32));
      float mnew = fmaxf(mrun[jv], v);
      alpha[jv] = __expf(mrun[jv] - mnew);
      mrun[jv] = mnew;
    }
    // p = exp(s - m); masked lanes forced to exact 0 (handles all-masked tiles)
#pragma unroll
    for (int jv = 0; jv < 8; ++jv) {
      p[0][jv] = (p[0][jv] <= -1.0e37f) ? 0.f : __expf(p[0][jv] - mrun[jv]);
      p[1][jv] = (p[1][jv] <= -1.0e37f) ? 0.f : __expf(p[1][jv] - mrun[jv]);
      float v = p[0][jv] + p[1][jv];
#pragma unroll
      for (int off = 8; off >= 1; off >>= 1) v += __shfl_xor(v, off, 32);
      lsum[jv] = lsum[jv] * alpha[jv] + v;
    }
#pragma unroll
    for (int nt = 0; nt < 4; ++nt)
#pragma unroll
      for (int jv = 0; jv < 8; ++jv) acc[nt][jv] *= alpha[jv];

    // P: C layout -> LDS (row major 16x32) -> reload in A layout
#pragma unroll
    for (int t = 0; t < 2; ++t)
#pragma unroll
      for (int jv = 0; jv < 8; ++jv)
        Ps[wv][(jv + half8) * 32 + t * 16 + lx] = (__bf16)p[t][jv];
    __asm__ volatile("s_wait_dscnt 0x0" ::: "memory");  // wave-local LDS RAW
    FragB pf;
    const __bf16* pr = &Ps[wv][lx * 32];
    pf.q[0] = *(const U4*)(pr + half8);
    pf.q[1] = *(const U4*)(pr + half8 + 16);

    // P @ V over 4 dim tiles of 16
#pragma unroll
    for (int nt = 0; nt < 4; ++nt) {
      FragB vf;
      int dcol = h * HD + nt * 16 + lx;
#pragma unroll
      for (int e = 0; e < 16; ++e) {
        int key = j0 + half8 + ((e < 8) ? e : e + 8);
        vf.v[e] = qkv[(size_t)(b * SS + key) * rowstride + 2 * DD + dcol];
      }
      acc[nt] = wmma_bf16(pf.v, vf.v, acc[nt]);
    }
  }
  // normalize + store bf16 for the Wo GEMM
  float invl[8];
#pragma unroll
  for (int jv = 0; jv < 8; ++jv) invl[jv] = 1.0f / lsum[jv];
#pragma unroll
  for (int nt = 0; nt < 4; ++nt)
#pragma unroll
    for (int jv = 0; jv < 8; ++jv) {
      int m = i0 + jv + half8;
      attn[(size_t)(b * SS + m) * DD + h * HD + nt * 16 + lx] =
          (__bf16)(acc[nt][jv] * invl[jv]);
    }
}

// -------------------------------------------- residual + LayerNorm (wave/row)
template <int HAS_RES>
__global__ __launch_bounds__(256) void ln_kernel(float* __restrict__ x,
                                                 const float* __restrict__ res,
                                                 const float* __restrict__ g,
                                                 const float* __restrict__ bt,
                                                 __bf16* __restrict__ xb) {
  int row = blockIdx.x * 8 + (threadIdx.x >> 5);
  int lane = threadIdx.x & 31;
  float vals[16];
  float s = 0.f, sq = 0.f;
#pragma unroll
  for (int k = 0; k < 16; ++k) {
    int d = lane + k * 32;
    float v = x[(size_t)row * DD + d];
    if (HAS_RES) v += res[(size_t)row * DD + d];
    vals[k] = v;
    s += v;
    sq += v * v;
  }
#pragma unroll
  for (int off = 16; off >= 1; off >>= 1) {
    s += __shfl_xor(s, off, 32);
    sq += __shfl_xor(sq, off, 32);
  }
  float mean = s * (1.f / DD);
  float var = sq * (1.f / DD) - mean * mean;  // biased, matches torch LN
  float rstd = rsqrtf(var + 1e-5f);
#pragma unroll
  for (int k = 0; k < 16; ++k) {
    int d = lane + k * 32;
    float y = (vals[k] - mean) * rstd * g[d] + bt[d];
    x[(size_t)row * DD + d] = y;
    xb[(size_t)row * DD + d] = (__bf16)y;
  }
}

// ------------------------------------------------------------------ SwiGLU
__global__ void swiglu_kernel(const float* __restrict__ hbuf,
                              __bf16* __restrict__ ff) {
  int i = blockIdx.x * blockDim.x + threadIdx.x;  // over 2048*2048
  int m = i >> 11, j = i & (FFN - 1);
  float u = hbuf[(size_t)m * (2 * FFN) + j];
  float gg = hbuf[(size_t)m * (2 * FFN) + FFN + j];
  float sig = 1.f / (1.f + __expf(-gg));
  ff[i] = (__bf16)(u * gg * sig);
}

// =================================================================== launcher
extern "C" void kernel_launch(void* const* d_in, const int* in_sizes, int n_in,
                              void* d_out, int out_size, void* d_ws, size_t ws_size,
                              hipStream_t stream) {
  (void)in_sizes; (void)n_in; (void)out_size; (void)ws_size;
  const int*   ids  = (const int*)d_in[0];
  const float* emb  = (const float*)d_in[1];
  const float* Wqkv = (const float*)d_in[2];
  const float* bqkv = (const float*)d_in[3];
  const float* Wo   = (const float*)d_in[4];
  const float* bo   = (const float*)d_in[5];
  const float* W1   = (const float*)d_in[6];
  const float* b1   = (const float*)d_in[7];
  const float* W2   = (const float*)d_in[8];
  const float* b2   = (const float*)d_in[9];
  const float* ln1g = (const float*)d_in[10];
  const float* ln1b = (const float*)d_in[11];
  const float* ln2g = (const float*)d_in[12];
  const float* ln2b = (const float*)d_in[13];
  const float* lnfg = (const float*)d_in[14];
  const float* lnfb = (const float*)d_in[15];

  char* p = (char*)d_ws;
  auto alloc = [&](size_t bytes) {
    char* r = p;
    p += (bytes + 255) & ~(size_t)255;
    return r;
  };
  __bf16* embB  = (__bf16*)alloc((size_t)VV * DD * 2);
  __bf16* wqkvB = (__bf16*)alloc((size_t)LL * 3 * DD * DD * 2);
  __bf16* woB   = (__bf16*)alloc((size_t)LL * DD * DD * 2);
  __bf16* w1B   = (__bf16*)alloc((size_t)LL * 2 * FFN * DD * 2);
  __bf16* w2B   = (__bf16*)alloc((size_t)LL * DD * FFN * 2);
  float*  x     = (float*)alloc((size_t)BB * SS * DD * 4);
  __bf16* xb    = (__bf16*)alloc((size_t)BB * SS * DD * 2);
  __bf16* qkvB  = (__bf16*)alloc((size_t)BB * SS * 3 * DD * 2);
  __bf16* attnB = (__bf16*)alloc((size_t)BB * SS * DD * 2);
  float*  tmp   = (float*)alloc((size_t)BB * SS * DD * 4);
  float*  hbuf  = (float*)alloc((size_t)BB * SS * 2 * FFN * 4);
  __bf16* ffB   = (__bf16*)alloc((size_t)BB * SS * FFN * 2);

  auto cvt = [&](const float* src, __bf16* dst, size_t n) {
    int blocks = (int)((n / 4 + 255) / 256);
    cvt_bf16_kernel<<<blocks, 256, 0, stream>>>(src, dst, (int)n);
  };
  cvt(emb, embB, (size_t)VV * DD);
  cvt(Wqkv, wqkvB, (size_t)LL * 3 * DD * DD);
  cvt(Wo, woB, (size_t)LL * DD * DD);
  cvt(W1, w1B, (size_t)LL * 2 * FFN * DD);
  cvt(W2, w2B, (size_t)LL * DD * FFN);

  embed_kernel<<<BB * SS, 256, 0, stream>>>(ids, emb, x, xb);

  const int Mrows = BB * SS;  // 2048
  for (int l = 0; l < LL; ++l) {
    gemm_kernel<1, 1><<<dim3(Mrows / 128, (3 * DD) / 64), 256, 0, stream>>>(
        xb, wqkvB + (size_t)l * 3 * DD * DD, bqkv + (size_t)l * 3 * DD, qkvB,
        Mrows, 3 * DD, DD);
    attn_kernel<<<dim3(SS / 16 / 4, HH, BB), 128, 0, stream>>>(qkvB, attnB);
    gemm_kernel<0, 1><<<dim3(Mrows / 128, DD / 64), 256, 0, stream>>>(
        attnB, woB + (size_t)l * DD * DD, bo + (size_t)l * DD, tmp, Mrows, DD, DD);
    ln_kernel<1><<<Mrows / 8, 256, 0, stream>>>(x, tmp, ln1g + (size_t)l * DD,
                                                ln1b + (size_t)l * DD, xb);
    gemm_kernel<0, 1><<<dim3(Mrows / 128, (2 * FFN) / 64), 256, 0, stream>>>(
        xb, w1B + (size_t)l * 2 * FFN * DD, b1 + (size_t)l * 2 * FFN, hbuf,
        Mrows, 2 * FFN, DD);
    swiglu_kernel<<<(Mrows * FFN) / 256, 256, 0, stream>>>(hbuf, ffB);
    gemm_kernel<0, 1><<<dim3(Mrows / 128, DD / 64), 256, 0, stream>>>(
        ffB, w2B + (size_t)l * DD * FFN, b2 + (size_t)l * DD, tmp, Mrows, DD, FFN);
    ln_kernel<1><<<Mrows / 8, 256, 0, stream>>>(x, tmp, ln2g + (size_t)l * DD,
                                                ln2b + (size_t)l * DD, xb);
  }
  ln_kernel<0><<<Mrows / 8, 256, 0, stream>>>(x, nullptr, lnfg, lnfb, xb);
  // tied LM head: [2048,512] @ [32000,512]^T -> d_out fp32
  gemm_kernel<0, 0><<<dim3(Mrows / 128, VV / 64), 256, 0, stream>>>(
      xb, embB, nullptr, d_out, Mrows, VV, DD);
}
